// MoE_82463372083941
// MI455X (gfx1250) — matrix operand order repaired and checked by reading, not compiled, not run on hardware
//
#include <hip/hip_runtime.h>
#include <hip/hip_bf16.h>
#include <math.h>

#define N_STATE   1024
#define N_CONCAT  2048
#define N_INTER   4096
#define N_EXPERTS 8
#define N_TOKENS  8192
#define LN_EPS    1e-5f

typedef __attribute__((ext_vector_type(16))) _Float16 v16h;
typedef __attribute__((ext_vector_type(8)))  float    v8f;
typedef int v4i_async __attribute__((vector_size(16)));

#define BM 64
#define BN 64
#define BK 32
#define LDF  36   // padded LDS row length for fp32 tiles (144B rows: conflict-free)
#define LDTH 40   // padded LDS row length for f16 tiles (80B rows)

union Frag16 { uint4 u[2]; v16h v; };
union Acc8   { v8f v; float f[8]; };

// ---------------------------------------------------------------------------
// CDNA5 async global->LDS copy (ASYNCcnt-tracked); fallback = sync copy.
// Builtin signature (from clang diagnostic): (v4i AS1* src, v4i AS3* dst, Ii, Ii)
// ---------------------------------------------------------------------------
#if __has_builtin(__builtin_amdgcn_global_load_async_to_lds_b128)
#define HAVE_ASYNC 1
#define ASYNC_CP16(gsrc, ldst)                                                  \
    __builtin_amdgcn_global_load_async_to_lds_b128(                             \
        (__attribute__((address_space(1))) v4i_async*)(gsrc),                   \
        (__attribute__((address_space(3))) v4i_async*)(ldst), 0, 0)
#else
#define HAVE_ASYNC 0
#define ASYNC_CP16(gsrc, ldst) (*(uint4*)(ldst) = *(const uint4*)(gsrc))
#endif

template<int N>
static __device__ __forceinline__ void wait_async() {
#if HAVE_ASYNC
    asm volatile("s_wait_asynccnt %0" :: "n"(N) : "memory");
#endif
}

static __device__ __forceinline__ float gelu_exact(float x) {
    return 0.5f * x * (1.0f + erff(x * 0.70710678118654752440f));
}

// Build a v16h fragment from two groups of 8 consecutive fp32 values in LDS.
static __device__ __forceinline__ v16h frag_from_f32(const float* p0, const float* p1) {
    union { _Float16 h[16]; v16h v; } t;
    float4 a = *(const float4*)p0;
    float4 b = *(const float4*)(p0 + 4);
    float4 c = *(const float4*)p1;
    float4 d = *(const float4*)(p1 + 4);
    t.h[0]=(_Float16)a.x;  t.h[1]=(_Float16)a.y;  t.h[2]=(_Float16)a.z;  t.h[3]=(_Float16)a.w;
    t.h[4]=(_Float16)b.x;  t.h[5]=(_Float16)b.y;  t.h[6]=(_Float16)b.z;  t.h[7]=(_Float16)b.w;
    t.h[8]=(_Float16)c.x;  t.h[9]=(_Float16)c.y;  t.h[10]=(_Float16)c.z; t.h[11]=(_Float16)c.w;
    t.h[12]=(_Float16)d.x; t.h[13]=(_Float16)d.y; t.h[14]=(_Float16)d.z; t.h[15]=(_Float16)d.w;
    return t.v;
}

// ---------------------------------------------------------------------------
// Init: zero Y accumulator and expert counters
// ---------------------------------------------------------------------------
__global__ void moe_init_kernel(float* __restrict__ Y, int* __restrict__ cnt) {
    long i = (long)blockIdx.x * blockDim.x + threadIdx.x;
    if (i < N_EXPERTS) cnt[i] = 0;
    const long total = (long)N_TOKENS * N_STATE;
    for (long j = i; j < total; j += (long)gridDim.x * blockDim.x) Y[j] = 0.0f;
}

// ---------------------------------------------------------------------------
// Gate: softmax over 8 experts, top-2, build per-expert token lists
// ---------------------------------------------------------------------------
__global__ __launch_bounds__(256)
void moe_gate_kernel(const float* __restrict__ x, const float* __restrict__ gw,
                     int* __restrict__ cnt, int* __restrict__ list,
                     float* __restrict__ lw) {
    const int t = blockIdx.x;
    const int w = threadIdx.x >> 5;
    const int lane = threadIdx.x & 31;
    __shared__ float sc[N_EXPERTS];

    const float* xr = x  + (long)t * N_CONCAT;
    const float* gr = gw + (long)w * N_CONCAT;
    float s = 0.0f;
    for (int k = lane; k < N_CONCAT; k += 32) s += xr[k] * gr[k];
    for (int o = 16; o > 0; o >>= 1) s += __shfl_xor(s, o, 32);
    if (lane == 0) sc[w] = s;
    __syncthreads();

    if (threadIdx.x == 0) {
        float m = sc[0];
        for (int e = 1; e < N_EXPERTS; ++e) m = fmaxf(m, sc[e]);
        float p[N_EXPERTS], den = 0.0f;
        for (int e = 0; e < N_EXPERTS; ++e) { p[e] = __expf(sc[e] - m); den += p[e]; }
        for (int e = 0; e < N_EXPERTS; ++e) p[e] /= den;
        int i0 = 0;
        for (int e = 1; e < N_EXPERTS; ++e) if (p[e] > p[i0]) i0 = e;
        int i1 = -1;
        for (int e = 0; e < N_EXPERTS; ++e) {
            if (e == i0) continue;
            if (i1 < 0 || p[e] > p[i1]) i1 = e;
        }
        int s0 = atomicAdd(&cnt[i0], 1);
        list[i0 * N_TOKENS + s0] = t;  lw[i0 * N_TOKENS + s0] = p[i0];
        int s1 = atomicAdd(&cnt[i1], 1);
        list[i1 * N_TOKENS + s1] = t;  lw[i1 * N_TOKENS + s1] = p[i1];
    }
}

// ---------------------------------------------------------------------------
// Shared-path bias fold: bs = sw2 @ sb1 + sb2
// ---------------------------------------------------------------------------
__global__ void moe_prep_bias_kernel(const float* __restrict__ sw2,
                                     const float* __restrict__ sb1,
                                     const float* __restrict__ sb2,
                                     float* __restrict__ bs) {
    int sIdx = blockIdx.x * blockDim.x + threadIdx.x;
    if (sIdx >= N_STATE) return;
    float a = 0.0f;
    const float* row = sw2 + (long)sIdx * N_INTER;
    for (int j = 0; j < N_INTER; ++j) a += row[j] * sb1[j];
    bs[sIdx] = a + sb2[sIdx];
}

// ---------------------------------------------------------------------------
// Tile staging: issue async global->LDS copies for one BMxBK / BNxBK pair.
//   MODE 0: A = gather(x,list) fp32 ; B = W1 fp32
//   MODE 1: A = H f16 (byte copy)   ; B = W2 fp32
//   MODE 2: A = x fp32              ; B = Ws fp32
//   MODE 3: A = sw2 fp32 (async)    ; B = sw1 transposed (manual stores)
// ---------------------------------------------------------------------------
template<int MODE>
static __device__ __forceinline__ void stage_tile(
    float* __restrict__ AsF, float* __restrict__ BsF,
    const float* __restrict__ Af, const unsigned short* __restrict__ Ah,
    const float* __restrict__ Bf, const int* __restrict__ sTok,
    int mBase0, int nBase0, int k0, int ldA, int ldB, int tid) {
    const int sr  = tid >> 2;        // row 0..63
    const int sk4 = (tid & 3) * 8;   // element chunk 0/8/16/24

    // ---- A tile
    if (MODE == 1) {
        const unsigned short* src = Ah + (long)(mBase0 + sr) * ldA + k0 + sk4;
        unsigned short* As16 = (unsigned short*)AsF;
        ASYNC_CP16(src, &As16[sr * LDTH + sk4]);          // 16B of f16, no conversion
    } else {
        long rowOff = (MODE == 0) ? (long)sTok[sr] * ldA : (long)(mBase0 + sr) * ldA;
        const float* src = Af + rowOff + k0 + sk4;
        ASYNC_CP16(src,     &AsF[sr * LDF + sk4]);
        ASYNC_CP16(src + 4, &AsF[sr * LDF + sk4 + 4]);
    }
    // ---- B tile
    if (MODE == 3) {
        const int n = nBase0 + sr;
        __builtin_prefetch(&Bf[(long)(k0 + sk4 + 8) * ldB + n], 0, 1);
#pragma unroll
        for (int i = 0; i < 8; ++i)
            BsF[sr * LDF + sk4 + i] = Bf[(long)(k0 + sk4 + i) * ldB + n];
    } else {
        const float* src = Bf + (long)(nBase0 + sr) * ldB + k0 + sk4;
        ASYNC_CP16(src,     &BsF[sr * LDF + sk4]);
        ASYNC_CP16(src + 4, &BsF[sr * LDF + sk4 + 4]);
    }
}

// ---------------------------------------------------------------------------
// Generic WMMA GEMM tile kernel: D(M,N) = op( A(M,K) * B(N,K)^T )
// 64x64 block tile, BK=32, 8 waves, double-buffered async LDS staging,
// 2 x v_wmma_f32_16x16x32_f16 per wave per K step.
// ---------------------------------------------------------------------------
template<int MODE>
__global__ __launch_bounds__(256)
void moe_gemm_kernel(const float* __restrict__ Af,
                     const unsigned short* __restrict__ Ah,
                     const float* __restrict__ Bf,
                     const float* __restrict__ bias,
                     const int* __restrict__ list,
                     const float* __restrict__ lw,
                     const int* __restrict__ cntPtr,
                     float* __restrict__ Cf,
                     unsigned short* __restrict__ Ch,
                     int Ktot, int ldA, int ldB, int ldC) {
    constexpr int BUFSZ  = BM * LDF + BN * LDF;               // floats per buffer
    constexpr int NASYNC = (MODE == 1) ? 3 : (MODE == 3) ? 2 : 4;  // asyncs/thread/tile
    __shared__ __align__(16) float smem[2 * BUFSZ];
    __shared__ int   sTok[BM];
    __shared__ float sWgt[BM];

    const int tid    = threadIdx.x;
    const int nBase0 = blockIdx.x * BN;
    const int mBase0 = blockIdx.y * BM;

    int cnt = 0;
    if (MODE == 0 || MODE == 1) {
        cnt = *cntPtr;                       // uniform
        if (mBase0 >= cnt) return;           // uniform early-exit (EXEC stays full)
        if (tid < BM) {
            int r = mBase0 + tid;
            int tok = 0; float wv = 0.0f;
            if (r < cnt) { tok = list[r]; if (MODE == 1) wv = lw[r]; }
            sTok[tid] = tok; sWgt[tid] = wv;
        }
        __syncthreads();
    }

    const int w    = tid >> 5;
    const int lane = tid & 31;
    const int half = lane >> 4;
    const int l16  = lane & 15;
    const int wm   = w & 3;          // M subtile (16 rows)
    const int wn   = w >> 2;         // N half (32 cols)

    v8f acc0 = {}; v8f acc1 = {};

    float* AsF0 = smem;              float* BsF0 = smem + BM * LDF;
    float* AsF1 = smem + BUFSZ;      float* BsF1 = smem + BUFSZ + BM * LDF;

    const int nKB = Ktot / BK;
    stage_tile<MODE>(AsF0, BsF0, Af, Ah, Bf, sTok, mBase0, nBase0, 0, ldA, ldB, tid);

    for (int kb = 0; kb < nKB; ++kb) {
        float* curA = (kb & 1) ? AsF1 : AsF0;
        float* curB = (kb & 1) ? BsF1 : BsF0;
        float* nxtA = (kb & 1) ? AsF0 : AsF1;
        float* nxtB = (kb & 1) ? BsF0 : BsF1;

        if (kb + 1 < nKB) {
            stage_tile<MODE>(nxtA, nxtB, Af, Ah, Bf, sTok,
                             mBase0, nBase0, (kb + 1) * BK, ldA, ldB, tid);
            wait_async<NASYNC>();    // previous batch complete (in-order)
        } else {
            wait_async<0>();
        }
        __syncthreads();

        // ---- fragments
        Frag16 af, bf0, bf1;
        if (MODE == 1) {
            const unsigned short* As16 = (const unsigned short*)curA;
            const unsigned short* ap = &As16[(wm * 16 + l16) * LDTH + half * 8];
            af.u[0] = *(const uint4*)ap;
            af.u[1] = *(const uint4*)(ap + 16);
        } else {
            const float* ap = &curA[(wm * 16 + l16) * LDF + half * 8];
            af.v = frag_from_f32(ap, ap + 16);
        }
        const float* bp0 = &curB[(wn * 32 + l16) * LDF + half * 8];
        bf0.v = frag_from_f32(bp0, bp0 + 16);
        const float* bp1 = &curB[(wn * 32 + 16 + l16) * LDF + half * 8];
        bf1.v = frag_from_f32(bp1, bp1 + 16);

        acc0 = __builtin_amdgcn_wmma_f32_16x16x32_f16(false, af.v, false, bf0.v,
                                                      (short)0, acc0, false, false);
        acc1 = __builtin_amdgcn_wmma_f32_16x16x32_f16(false, af.v, false, bf1.v,
                                                      (short)0, acc1, false, false);
        __syncthreads();             // all waves done with cur before it is restaged
    }

    // ---- epilogue
    Acc8 a0, a1; a0.v = acc0; a1.v = acc1;
#pragma unroll
    for (int i = 0; i < 8; ++i) {
        const int m  = wm * 16 + i + half * 8;
        const int n0 = nBase0 + wn * 32 + l16;
        const int n1 = n0 + 16;

        if (MODE == 0) {
            int r = mBase0 + m;
            if (r < cnt) {
                float h0 = gelu_exact(a0.f[i] + bias[n0]);
                float h1 = gelu_exact(a1.f[i] + bias[n1]);
                union { _Float16 h; unsigned short u; } c0, c1;
                c0.h = (_Float16)h0; c1.h = (_Float16)h1;
                Ch[(long)r * ldC + n0] = c0.u;
                Ch[(long)r * ldC + n1] = c1.u;
            }
        } else if (MODE == 1) {
            int r = mBase0 + m;
            if (r < cnt) {
                int   tok = sTok[m];
                float wv  = sWgt[m];
                float* yr = Cf + (long)tok * ldC;
                yr[n0] += wv * (a0.f[i] + bias[n0]);
                yr[n1] += wv * (a1.f[i] + bias[n1]);
            }
        } else if (MODE == 2) {
            float* yr = Cf + (long)(mBase0 + m) * ldC;
            yr[n0] += a0.f[i] + bias[n0];
            yr[n1] += a1.f[i] + bias[n1];
        } else { // MODE == 3
            float* dr = Cf + (long)(mBase0 + m) * ldC;
            dr[n0] = a0.f[i];
            dr[n1] = a1.f[i];
        }
    }
}

// ---------------------------------------------------------------------------
// LayerNorm over N_STATE, out = LN(Y * 0.5) * g + b
// ---------------------------------------------------------------------------
__global__ __launch_bounds__(256)
void moe_ln_kernel(const float* __restrict__ Y, const float* __restrict__ g,
                   const float* __restrict__ b, float* __restrict__ out) {
    const int t   = blockIdx.x;
    const int tid = threadIdx.x;
    __shared__ float ssum[8], ssq[8];

    float v[4], s = 0.0f, sq = 0.0f;
    const float* yr = Y + (long)t * N_STATE;
#pragma unroll
    for (int i = 0; i < 4; ++i) {
        float xv = yr[tid + 256 * i] * 0.5f;
        v[i] = xv; s += xv; sq += xv * xv;
    }
    for (int o = 16; o > 0; o >>= 1) { s += __shfl_xor(s, o, 32); sq += __shfl_xor(sq, o, 32); }
    const int w = tid >> 5, lane = tid & 31;
    if (lane == 0) { ssum[w] = s; ssq[w] = sq; }
    __syncthreads();
    if (w == 0) {
        s  = (lane < 8) ? ssum[lane] : 0.0f;
        sq = (lane < 8) ? ssq[lane]  : 0.0f;
        for (int o = 4; o > 0; o >>= 1) { s += __shfl_xor(s, o, 32); sq += __shfl_xor(sq, o, 32); }
        if (lane == 0) { ssum[0] = s; ssq[0] = sq; }
    }
    __syncthreads();
    const float mean = ssum[0] * (1.0f / N_STATE);
    const float var  = ssq[0] * (1.0f / N_STATE) - mean * mean;
    const float rstd = rsqrtf(var + LN_EPS);
    float* orow = out + (long)t * N_STATE;
#pragma unroll
    for (int i = 0; i < 4; ++i) {
        int c = tid + 256 * i;
        orow[c] = (v[i] - mean) * rstd * g[c] + b[c];
    }
}

// ---------------------------------------------------------------------------
extern "C" void kernel_launch(void* const* d_in, const int* in_sizes, int n_in,
                              void* d_out, int out_size, void* d_ws, size_t ws_size,
                              hipStream_t stream) {
    (void)in_sizes; (void)n_in; (void)out_size; (void)ws_size;
    const float* x      = (const float*)d_in[0];
    const float* gate_w = (const float*)d_in[1];
    const float* ew1    = (const float*)d_in[2];
    const float* eb1    = (const float*)d_in[3];
    const float* ew2    = (const float*)d_in[4];
    const float* eb2    = (const float*)d_in[5];
    const float* sw1    = (const float*)d_in[6];
    const float* sb1    = (const float*)d_in[7];
    const float* sw2    = (const float*)d_in[8];
    const float* sb2    = (const float*)d_in[9];
    const float* ln_g   = (const float*)d_in[10];
    const float* ln_b   = (const float*)d_in[11];
    float* out = (float*)d_out;

    // workspace layout (all 256B aligned)
    char* ws = (char*)d_ws;
    size_t off = 0;
    int*   cnt  = (int*)(ws + off);   off += 256;
    int*   list = (int*)(ws + off);   off += (size_t)N_EXPERTS * N_TOKENS * sizeof(int);
    float* lw   = (float*)(ws + off); off += (size_t)N_EXPERTS * N_TOKENS * sizeof(float);
    float* bs   = (float*)(ws + off); off += 4096;
    float* Ws   = (float*)(ws + off); off += (size_t)N_STATE * N_CONCAT * sizeof(float);   // 8 MB
    float* Y    = (float*)(ws + off); off += (size_t)N_TOKENS * N_STATE * sizeof(float);   // 32 MB
    unsigned short* H = (unsigned short*)(ws + off);                                       // 64 MB

    moe_init_kernel<<<2048, 256, 0, stream>>>(Y, cnt);
    moe_gate_kernel<<<N_TOKENS, 256, 0, stream>>>(x, gate_w, cnt, list, lw);
    moe_prep_bias_kernel<<<4, 256, 0, stream>>>(sw2, sb1, sb2, bs);

    // Ws = sw2 @ sw1  (M=1024, N=2048, K=4096); B accessed transposed
    moe_gemm_kernel<3><<<dim3(N_CONCAT / BN, N_STATE / BM), 256, 0, stream>>>(
        sw2, nullptr, sw1, nullptr, nullptr, nullptr, nullptr,
        Ws, nullptr, N_INTER, N_INTER, N_CONCAT, N_CONCAT);

    for (int e = 0; e < N_EXPERTS; ++e) {
        const float* w1 = ew1 + (size_t)e * N_INTER * N_CONCAT;
        const float* b1 = eb1 + (size_t)e * N_INTER;
        const float* w2 = ew2 + (size_t)e * N_STATE * N_INTER;
        const float* b2 = eb2 + (size_t)e * N_STATE;
        // H = GELU(gather(x) @ W1^T + b1): M<=8192 (routed), N=4096, K=2048
        moe_gemm_kernel<0><<<dim3(N_INTER / BN, N_TOKENS / BM), 256, 0, stream>>>(
            x, nullptr, w1, b1, list + e * N_TOKENS, nullptr, cnt + e,
            nullptr, H, N_CONCAT, N_CONCAT, N_CONCAT, N_INTER);
        // Y[token] += w * (H @ W2^T + b2): N=1024, K=4096
        moe_gemm_kernel<1><<<dim3(N_STATE / BN, N_TOKENS / BM), 256, 0, stream>>>(
            nullptr, H, w2, b2, list + e * N_TOKENS, lw + e * N_TOKENS, cnt + e,
            Y, nullptr, N_INTER, N_INTER, N_INTER, N_STATE);
    }

    // shared path: Y += x @ Ws^T + bs  (M=8192, N=1024, K=2048)
    moe_gemm_kernel<2><<<dim3(N_STATE / BN, N_TOKENS / BM), 256, 0, stream>>>(
        x, nullptr, Ws, bs, nullptr, nullptr, nullptr,
        Y, nullptr, N_CONCAT, N_CONCAT, N_CONCAT, N_STATE);

    moe_ln_kernel<<<N_TOKENS, 256, 0, stream>>>(Y, ln_g, ln_b, out);
}
